// SlotAttention_70884140253475
// MI455X (gfx1250) — matrix operand order repaired and checked by reading, compiled, not run on hardware
//
#include <hip/hip_runtime.h>
#include <hip/hip_bf16.h>

typedef __bf16 bf16;
typedef __attribute__((ext_vector_type(16))) __bf16 v16bf;
typedef __attribute__((ext_vector_type(8)))  __bf16 v8bf;
typedef __attribute__((ext_vector_type(8)))  float  v8f;

#define LN_EPS   1e-5f
#define ATTN_EPS 1e-8f

__device__ __forceinline__ v8f wmma_bf(v16bf a, v16bf b, v8f c) {
  // v_wmma_f32_16x16x32_bf16
  return __builtin_amdgcn_wmma_f32_16x16x32_bf16(false, a, false, b, (short)0, c,
                                                 false, false);
}

// A-fragment 16x32 (MxK), row-major source with leading dim `ld` (elements).
// ISA layout: lanes 0-15 -> rows, K = [k0..k0+7] and [k0+16..k0+23];
// lanes 16-31 same rows, K shifted by +8.
__device__ __forceinline__ v16bf load_a16(const bf16* p, int ld, int row0, int k0) {
  int lane = threadIdx.x & 31;
  const bf16* r = p + (size_t)(row0 + (lane & 15)) * ld + k0 + ((lane >> 4) << 3);
  union { v16bf v; v8bf h[2]; } u;
  u.h[0] = *(const v8bf*)(r);
  u.h[1] = *(const v8bf*)(r + 16);
  return u.v;
}

// B-fragment 32x16 (KxN) sourced from row-major W[N,K] (computes X @ W^T).
// Lane = output column; lanes 0-15 K=[k0..k0+7],[k0+16..k0+23]; lanes 16-31 K shifted by +8.
__device__ __forceinline__ v16bf load_bt16(const bf16* W, int ld, int j0, int k0) {
  int lane = threadIdx.x & 31;
  const bf16* r = W + (size_t)(j0 + (lane & 15)) * ld + k0 + ((lane >> 4) << 3);
  union { v16bf v; v8bf h[2]; } u;
  u.h[0] = *(const v8bf*)(r);
  u.h[1] = *(const v8bf*)(r + 16);
  return u.v;
}

__device__ __forceinline__ float wred(float v) {   // wave32 sum
  v += __shfl_xor(v, 16, 32);
  v += __shfl_xor(v, 8, 32);
  v += __shfl_xor(v, 4, 32);
  v += __shfl_xor(v, 2, 32);
  v += __shfl_xor(v, 1, 32);
  return v;
}

// LayerNorm one 256-elem row per wave, write bf16.
__device__ __forceinline__ void ln256_row(const float* src, bf16* dst,
                                          const float* gamma, const float* beta) {
  int lane = threadIdx.x & 31;
  float x[8]; float s = 0.f, s2 = 0.f;
#pragma unroll
  for (int j = 0; j < 8; ++j) { float t = src[lane*8+j]; x[j]=t; s+=t; s2+=t*t; }
  s = wred(s); s2 = wred(s2);
  float m  = s * (1.0f/256.0f);
  float var = s2 * (1.0f/256.0f) - m*m;
  float rs = rsqrtf(var + LN_EPS);
#pragma unroll
  for (int j = 0; j < 8; ++j)
    dst[lane*8+j] = (bf16)((x[j]-m)*rs*gamma[lane*8+j] + beta[lane*8+j]);
}

__global__ void cvt_kernel(const float* __restrict__ src, bf16* __restrict__ dst, int n) {
  int i = blockIdx.x * 256 + threadIdx.x;
  if (i < n) dst[i] = (bf16)src[i];
}

// ------------------------------------------------------------------
// Kernel: LN(inputs) then k = ln(x)@Wk^T * 1/16 (bf16), vT = (ln(x)@Wv^T)^T (bf16)
// 64 rows per block of the flattened [B*N, 256] input.
// ------------------------------------------------------------------
__global__ __launch_bounds__(256) void kv_kernel(
    const float* __restrict__ X, const float* __restrict__ g_in,
    const float* __restrict__ b_in, const bf16* __restrict__ Wk,
    const bf16* __restrict__ Wv, bf16* __restrict__ Kout, bf16* __restrict__ VTout) {
  __shared__ __align__(16) bf16 xln[64 * 272];          // 34.8 KB, padded stride
  const int tid = threadIdx.x, lane = tid & 31, wave = tid >> 5;
  const size_t rowbase = (size_t)blockIdx.x * 64;

  // Phase 1: LayerNorm, 8 rows per wave
  for (int r = 0; r < 8; ++r) {
    int row = wave * 8 + r;
    ln256_row(X + (rowbase + row) * 256, xln + row * 272, g_in, b_in);
  }
  __syncthreads();

  // Phase 2: WMMA GEMM. 2 waves per 16-row strip; each wave 8 column tiles.
  const int strip = wave >> 1, half = wave & 1;
  const int m0 = strip * 16;
  const int scol = lane & 15, hi = lane >> 4;
  v16bf afrag[8];
#pragma unroll
  for (int kk = 0; kk < 8; ++kk) afrag[kk] = load_a16(xln, 272, m0, kk * 32);

  const size_t b = rowbase >> 12;                        // batch index
  const int nbase = (int)(rowbase & 4095) + m0 + hi * 8; // n within batch
#pragma unroll 1
  for (int i = 0; i < 8; ++i) {
    int j0 = (half + 2 * i) * 16;
    v8f acck = {}, accv = {};
#pragma unroll
    for (int kk = 0; kk < 8; ++kk) {
      acck = wmma_bf(afrag[kk], load_bt16(Wk, 256, j0, kk * 32), acck);
      accv = wmma_bf(afrag[kk], load_bt16(Wv, 256, j0, kk * 32), accv);
    }
    int col = j0 + scol;
    size_t krow = rowbase + m0 + hi * 8;
    bf16* vp = VTout + ((b * 256 + col) * 4096 + nbase);
#pragma unroll
    for (int r = 0; r < 8; ++r) {
      Kout[(krow + r) * 256 + col] = (bf16)(acck[r] * 0.0625f); // * S^-0.5
      vp[r] = (bf16)accv[r];                                    // transposed v
    }
  }
}

// ------------------------------------------------------------------
// Kernel: full 3-iteration slot-attention loop; one block per batch.
// LDS phase-aliased pool (< 64 KB).
// ------------------------------------------------------------------
__global__ __launch_bounds__(256) void iter_kernel(
    const float* __restrict__ noise, const float* __restrict__ slot_mu,
    const float* __restrict__ slot_ls, const float* __restrict__ g_sl,
    const float* __restrict__ b_sl, const float* __restrict__ g_mlp,
    const float* __restrict__ b_mlp, const float* __restrict__ b_ih,
    const float* __restrict__ b_hh, const float* __restrict__ b1,
    const float* __restrict__ b2,
    const bf16* __restrict__ kbf, const bf16* __restrict__ vT,
    bf16* __restrict__ attn_ws,
    const bf16* __restrict__ Wq, const bf16* __restrict__ Wih,
    const bf16* __restrict__ Whh, const bf16* __restrict__ W1,
    const bf16* __restrict__ W2, float* __restrict__ out) {
  __shared__ __align__(16) char smem[60544];
  float* slots_f = (float*)smem;                 // [8][256] fp32 slots
  bf16*  a_bf    = (bf16*)(smem + 8192);         // [16][272]: ln / updates
  bf16*  b_bf    = (bf16*)(smem + 16896);        // [16][272]: q / slots(bf16)
  bf16*  gr      = (bf16*)(smem + 25600);        // gate buffers [16][272] each
  bf16*  gz      = gr   + 16 * 272;
  bf16*  gni     = gz   + 16 * 272;
  bf16*  gnh     = gni  + 16 * 272;
  bf16*  h_bf    = (bf16*)(smem + 25600);        // [16][528] MLP hidden (aliases gates)
  float* colsum  = (float*)(smem + 60416);
  float* invsum  = (float*)(smem + 60480);

  const int tid = threadIdx.x, lane = tid & 31, wave = tid >> 5;
  const int scol = lane & 15, hi = lane >> 4;
  const int b = blockIdx.x;
  const bf16* kB = kbf + (size_t)b * 4096 * 256;
  const bf16* vB = vT  + (size_t)b * 256 * 4096;
  bf16* attnB    = attn_ws + (size_t)b * 16 * 4096;

  // slots = mu + exp(log_sigma) * noise
  const float* nz = noise + (size_t)b * 2048;
  for (int t = 0; t < 8; ++t) {
    int idx = tid + t * 256;
    slots_f[idx] = slot_mu[idx & 255] + __expf(slot_ls[idx & 255]) * nz[idx];
  }
  __syncthreads();

  for (int it = 0; it < 3; ++it) {
    // (a) ln(slots) with g_sl -> a_bf
    ln256_row(slots_f + wave * 256, a_bf + wave * 272, g_sl, b_sl);
    __syncthreads();

    // (b) q = ln @ Wq^T -> b_bf[slot][s]
#pragma unroll
    for (int i = 0; i < 2; ++i) {
      int j0 = (wave * 2 + i) * 16;
      v8f acc = {};
#pragma unroll
      for (int kk = 0; kk < 8; ++kk)
        acc = wmma_bf(load_a16(a_bf, 272, 0, kk * 32),
                      load_bt16(Wq, 256, j0, kk * 32), acc);
      if (lane < 16) {
        int col = j0 + lane;
#pragma unroll
        for (int r = 0; r < 8; ++r) b_bf[r * 272 + col] = (bf16)acc[r];
      }
    }
    if (tid < 16) colsum[tid] = 0.f;
    __syncthreads();

    // (c) logits = k.q, softmax over 8 slots, attn (+EPS) -> ws, colsum.
    // q B-fragments are loop-invariant: hoist all 8 into registers.
    v16bf qfrag[8];
#pragma unroll
    for (int kk = 0; kk < 8; ++kk) qfrag[kk] = load_bt16(b_bf, 272, 0, kk * 32);
    float csum = 0.f;
#pragma unroll 1
    for (int tile = wave; tile < 256; tile += 8) {
      int n0 = tile * 16;
      v8f acc = {};
#pragma unroll
      for (int kk = 0; kk < 8; ++kk)
        acc = wmma_bf(load_a16(kB, 256, n0, kk * 32), qfrag[kk], acc);
#pragma unroll
      for (int r = 0; r < 8; ++r) {
        float v = acc[r];
        float mx = v;                       // reduce over 8-lane slot groups
        mx = fmaxf(mx, __shfl_xor(mx, 1, 32));
        mx = fmaxf(mx, __shfl_xor(mx, 2, 32));
        mx = fmaxf(mx, __shfl_xor(mx, 4, 32));
        float e = __expf(v - mx);
        float se = e;
        se += __shfl_xor(se, 1, 32);
        se += __shfl_xor(se, 2, 32);
        se += __shfl_xor(se, 4, 32);
        float a = e / se + ATTN_EPS;
        if (scol < 8) {
          attnB[(size_t)scol * 4096 + (n0 + r + hi * 8)] = (bf16)a;
          csum += a;
        }
      }
    }
    csum += __shfl_xor(csum, 16, 32);
    if (lane < 8) atomicAdd(&colsum[lane], csum);
    __syncthreads();
    if (tid < 8) invsum[tid] = 1.0f / colsum[tid];
    __syncthreads();

    // (d) updates^T = vT x attn, scaled by 1/colsum -> a_bf[slot][s]
    {
      int s0 = wave * 32;
      v8f acc0 = {}, acc1 = {};
#pragma unroll 2
      for (int kk = 0; kk < 128; ++kk) {
        if (kk + 4 < 128) {   // stay ahead of the WMMA chain on the vT stream
          const bf16* pf0 = vB + (size_t)(s0 + scol) * 4096 + (kk + 4) * 32 + (hi << 3);
          const bf16* pf1 = pf0 + (size_t)16 * 4096;
          __builtin_prefetch(pf0, 0, 0);
          __builtin_prefetch(pf1, 0, 0);
        }
        v16bf bb = load_bt16(attnB, 4096, 0, kk * 32);
        acc0 = wmma_bf(load_a16(vB, 4096, s0,      kk * 32), bb, acc0);
        acc1 = wmma_bf(load_a16(vB, 4096, s0 + 16, kk * 32), bb, acc1);
      }
      if (scol < 8) {
        float inv = invsum[scol];
#pragma unroll
        for (int r = 0; r < 8; ++r) {
          a_bf[scol * 272 + s0 + r + hi * 8]      = (bf16)(acc0[r] * inv);
          a_bf[scol * 272 + s0 + 16 + r + hi * 8] = (bf16)(acc1[r] * inv);
        }
      }
    }
    __syncthreads();

    // (e) GRU: slots(bf16) copy, gi/gh GEMMs, gate combine
    for (int t = 0; t < 8; ++t) {
      int idx = tid + t * 256;
      b_bf[(idx >> 8) * 272 + (idx & 255)] = (bf16)slots_f[idx];
    }
    __syncthreads();
    {
      // A-fragments (updates, slots) are invariant across the 6 column tiles.
      v16bf ufrag[8], sfrag[8];
#pragma unroll
      for (int kk = 0; kk < 8; ++kk) {
        ufrag[kk] = load_a16(a_bf, 272, 0, kk * 32);
        sfrag[kk] = load_a16(b_bf, 272, 0, kk * 32);
      }
#pragma unroll 1
      for (int i = 0; i < 6; ++i) {
        int j0 = (i * 8 + wave) * 16;
        v8f ai = {}, ah = {};
#pragma unroll
        for (int kk = 0; kk < 8; ++kk) {
          ai = wmma_bf(ufrag[kk], load_bt16(Wih, 256, j0, kk * 32), ai);
          ah = wmma_bf(sfrag[kk], load_bt16(Whh, 256, j0, kk * 32), ah);
        }
        if (lane < 16) {
          int jcol = j0 + lane, sec = j0 >> 8, sc = jcol & 255;
          float bi = b_ih[jcol], bh = b_hh[jcol];
#pragma unroll
          for (int r = 0; r < 8; ++r) {
            float gi = ai[r] + bi, gh = ah[r] + bh;
            if (sec == 0)      gr[r * 272 + sc] = (bf16)(gi + gh);
            else if (sec == 1) gz[r * 272 + sc] = (bf16)(gi + gh);
            else { gni[r * 272 + sc] = (bf16)gi; gnh[r * 272 + sc] = (bf16)gh; }
          }
        }
      }
    }
    __syncthreads();
    for (int t = 0; t < 8; ++t) {
      int idx = tid + t * 256;
      int ks = idx >> 8, s = idx & 255;
      float rg = 1.0f / (1.0f + __expf(-(float)gr[ks * 272 + s]));
      float zg = 1.0f / (1.0f + __expf(-(float)gz[ks * 272 + s]));
      float ng = tanhf((float)gni[ks * 272 + s] + rg * (float)gnh[ks * 272 + s]);
      float h = slots_f[idx];
      slots_f[idx] = (1.0f - zg) * ng + zg * h;
    }
    __syncthreads();

    // (f) MLP residual (iters 0,1)
    if (it < 2) {
      ln256_row(slots_f + wave * 256, a_bf + wave * 272, g_mlp, b_mlp);
      __syncthreads();
      {
        v16bf lfrag[8];
#pragma unroll
        for (int kk = 0; kk < 8; ++kk) lfrag[kk] = load_a16(a_bf, 272, 0, kk * 32);
#pragma unroll 1
        for (int i = 0; i < 4; ++i) {
          int j0 = (i * 8 + wave) * 16;
          v8f acc = {};
#pragma unroll
          for (int kk = 0; kk < 8; ++kk)
            acc = wmma_bf(lfrag[kk], load_bt16(W1, 256, j0, kk * 32), acc);
          if (lane < 16) {
            int jcol = j0 + lane;
            float bb = b1[jcol];
#pragma unroll
            for (int r = 0; r < 8; ++r)
              h_bf[r * 528 + jcol] = (bf16)fmaxf(acc[r] + bb, 0.0f);
          }
        }
      }
      __syncthreads();
#pragma unroll 1
      for (int i = 0; i < 2; ++i) {
        int j0 = (i * 8 + wave) * 16;
        v8f acc = {};
#pragma unroll
        for (int kk = 0; kk < 16; ++kk)
          acc = wmma_bf(load_a16(h_bf, 528, 0, kk * 32),
                        load_bt16(W2, 512, j0, kk * 32), acc);
        if (lane < 16) {
          int jcol = j0 + lane;
          float bb = b2[jcol];
#pragma unroll
          for (int r = 0; r < 8; ++r) slots_f[r * 256 + jcol] += acc[r] + bb;
        }
      }
      __syncthreads();
    }
  }

  float* ob = out + (size_t)b * 2048;
  for (int t = 0; t < 8; ++t) { int idx = tid + t * 256; ob[idx] = slots_f[idx]; }
}

extern "C" void kernel_launch(void* const* d_in, const int* in_sizes, int n_in,
                              void* d_out, int out_size, void* d_ws, size_t ws_size,
                              hipStream_t stream) {
  (void)in_sizes; (void)n_in; (void)out_size; (void)ws_size;
  const float* inputs  = (const float*)d_in[0];
  const float* noise   = (const float*)d_in[1];
  const float* slot_mu = (const float*)d_in[2];
  const float* slot_ls = (const float*)d_in[3];
  const float* g_in    = (const float*)d_in[4];
  const float* b_in    = (const float*)d_in[5];
  const float* g_sl    = (const float*)d_in[6];
  const float* b_sl    = (const float*)d_in[7];
  const float* g_mlp   = (const float*)d_in[8];
  const float* b_mlp   = (const float*)d_in[9];
  const float* Wq      = (const float*)d_in[10];
  const float* Wk      = (const float*)d_in[11];
  const float* Wv      = (const float*)d_in[12];
  const float* Wih     = (const float*)d_in[13];
  const float* Whh     = (const float*)d_in[14];
  const float* b_ih    = (const float*)d_in[15];
  const float* b_hh    = (const float*)d_in[16];
  const float* W1      = (const float*)d_in[17];
  const float* b1      = (const float*)d_in[18];
  const float* W2      = (const float*)d_in[19];
  const float* b2      = (const float*)d_in[20];

  // workspace carve-out (bf16)
  const size_t KV = (size_t)64 * 4096 * 256;
  bf16* kbf  = (bf16*)d_ws;
  bf16* vT   = kbf + KV;
  bf16* attn = vT + KV;
  bf16* wq   = attn + (size_t)64 * 16 * 4096;
  bf16* wk   = wq  + 256 * 256;
  bf16* wv   = wk  + 256 * 256;
  bf16* wih  = wv  + 256 * 256;
  bf16* whh  = wih + 768 * 256;
  bf16* w1   = whh + 768 * 256;
  bf16* w2   = w1  + 512 * 256;

  cvt_kernel<<<256, 256, 0, stream>>>(Wq,  wq,  65536);
  cvt_kernel<<<256, 256, 0, stream>>>(Wk,  wk,  65536);
  cvt_kernel<<<256, 256, 0, stream>>>(Wv,  wv,  65536);
  cvt_kernel<<<768, 256, 0, stream>>>(Wih, wih, 196608);
  cvt_kernel<<<768, 256, 0, stream>>>(Whh, whh, 196608);
  cvt_kernel<<<512, 256, 0, stream>>>(W1,  w1,  131072);
  cvt_kernel<<<512, 256, 0, stream>>>(W2,  w2,  131072);

  kv_kernel<<<4096, 256, 0, stream>>>(inputs, g_in, b_in, wk, wv, kbf, vT);
  iter_kernel<<<64, 256, 0, stream>>>(noise, slot_mu, slot_ls, g_sl, b_sl,
      g_mlp, b_mlp, b_ih, b_hh, b1, b2, kbf, vT, attn, wq, wih, whh, w1, w2,
      (float*)d_out);
}
